// RDN_56564719288333
// MI455X (gfx1250) — 8-bit matrix operand order repaired and checked by reading, compiled, not run on hardware
//
#include <hip/hip_runtime.h>
#include <stdint.h>
#include <stddef.h>


typedef __attribute__((ext_vector_type(16))) int   v16i;
typedef __attribute__((ext_vector_type(8)))  float v8f;

#define NPIX 8192
#define NSLOT 8448
#define EPS_BN 1e-5f

__device__ __forceinline__ unsigned char q8(float v) {
  return v > 0.f ? (unsigned char)0x38 : (v < 0.f ? (unsigned char)0xB8
                                                  : (unsigned char)0x00);
}
__device__ __forceinline__ int qperm(int c) {
  int o = c & 63;
  return (c & ~63) | (((o >> 3) & 1) << 5) | (((o >> 4) & 3) << 3) | (o & 7);
}
__device__ __forceinline__ int qslot(int p) {
  return (p >> 6) * 66 + (p & 63) + 1;
}

__global__ void pack_weights_kernel(const float* __restrict__ w,
                                    unsigned char* __restrict__ pack,
                                    int cin, int taps, int cChunks, int pairs) {
  int tid = blockIdx.x * blockDim.x + threadIdx.x;
  int total = taps * pairs * 4 * 2048;
  if (tid >= total) return;
  int b    = tid & 3;
  int j    = (tid >> 2) & 15;
  int lane = (tid >> 6) & 31;
  int tile = tid >> 11;
  int nt   = tile & 3;
  int kt   = tile >> 2;
  int tap  = kt / pairs;
  int pr   = kt - tap * pairs;
  int half = j >> 3, jj = j & 7;
  int cc   = pr * 2 + half;
  unsigned char q = 0;
  if (cc < cChunks) {
    int K64 = ((jj >> 1) * 16) + ((lane >> 4) * 8) + ((jj & 1) * 4) + b;
    int ci  = cc * 64 + K64;
    int kh = (taps == 9) ? (tap / 3) : 0;
    int kw = (taps == 9) ? (tap % 3) : 0;
    int ks = (taps == 9) ? 3 : 1;
    int N  = nt * 16 + (lane & 15);
    float wv = w[(((size_t)N * cin + ci) * ks + kh) * ks + kw];
    q = q8(wv);
  }
  pack[tid] = q;
}

__global__ void zero_words(unsigned int* __restrict__ dst, int nWords) {
  int t = blockIdx.x * blockDim.x + threadIdx.x;
  if (t < nWords) dst[t] = 0u;
}

template <int TAPS>
__global__ __launch_bounds__(256)
void bgemm_fp8(const unsigned char* __restrict__ Aq, int aStride,
               int cChunks,
               const unsigned char* __restrict__ Bpack,
               const float* __restrict__ bnG, const float* __restrict__ bnB,
               const float* __restrict__ bnM, const float* __restrict__ bnV,
               const float* __restrict__ res0, int r0Stride, int r0Off,
               const float* __restrict__ res1, int r1Stride, int r1Off,
               int doLeaky,
               float* __restrict__ outF, int oStride, int oOff,
               unsigned char* __restrict__ outQ, int qStride, int qOff) {
  const int lane = threadIdx.x & 31;
  const int wave = threadIdx.x >> 5;
  const int mt = wave & 3;
  const int np = wave >> 2;
  const int pixBase = blockIdx.x * 64 + mt * 16;
  const int m  = lane & 15;
  const int hi = lane >> 4;
  const int y  = (pixBase >> 6) & 63;
  const int psL = (pixBase >> 6) * 66 + (pixBase & 63) + 1 + m;

  v8f acc0 = {}, acc1 = {};
  const int pairs     = (cChunks + 1) >> 1;
  const int fullPairs = cChunks >> 1;
  const unsigned char* Bl = Bpack + (size_t)(np * 2) * 2048 + (size_t)lane * 64;

#pragma unroll
  for (int tap = 0; tap < TAPS; ++tap) {
    const int dy = (TAPS == 9) ? (tap / 3 - 1) : 0;
    const int dx = (TAPS == 9) ? (tap % 3 - 1) : 0;
    if (TAPS == 9) {
      const int yy = y + dy;
      if ((unsigned)yy >= 64u) continue;
    }
    const unsigned char* arow =
        Aq + (ptrdiff_t)(psL + dy * 66 + dx) * aStride + hi * 32;
    const unsigned char* Bt = Bl + (size_t)(tap * pairs) * 4 * 2048;

#pragma unroll 2
    for (int pr = 0; pr < fullPairs; ++pr) {
      const int* ai = (const int*)(arow + (size_t)pr * 128);
      v16i a;
#pragma unroll
      for (int k = 0; k < 8; ++k) a[k] = ai[k];
#pragma unroll
      for (int k = 0; k < 8; ++k) a[8 + k] = ai[16 + k];
      const int* bp0 = (const int*)(Bt + (size_t)(pr * 4) * 2048);
      const int* bp1 = bp0 + 512;
      v16i b0, b1;
#pragma unroll
      for (int k = 0; k < 16; ++k) b0[k] = bp0[k];
#pragma unroll
      for (int k = 0; k < 16; ++k) b1[k] = bp1[k];
      acc0 = __builtin_amdgcn_wmma_f32_16x16x128_fp8_fp8(a, b0, (short)0, acc0,
                                                         false, false);
      acc1 = __builtin_amdgcn_wmma_f32_16x16x128_fp8_fp8(a, b1, (short)0, acc1,
                                                         false, false);
    }
    if (cChunks & 1) {
      const int* ai = (const int*)(arow + (size_t)fullPairs * 128);
      v16i a;
#pragma unroll
      for (int k = 0; k < 8; ++k) a[k] = ai[k];
#pragma unroll
      for (int k = 0; k < 8; ++k) a[8 + k] = 0;
      const int* bp0 = (const int*)(Bt + (size_t)(fullPairs * 4) * 2048);
      const int* bp1 = bp0 + 512;
      v16i b0, b1;
#pragma unroll
      for (int k = 0; k < 16; ++k) b0[k] = bp0[k];
#pragma unroll
      for (int k = 0; k < 16; ++k) b1[k] = bp1[k];
      acc0 = __builtin_amdgcn_wmma_f32_16x16x128_fp8_fp8(a, b0, (short)0, acc0,
                                                         false, false);
      acc1 = __builtin_amdgcn_wmma_f32_16x16x128_fp8_fp8(a, b1, (short)0, acc1,
                                                         false, false);
    }
  }

  const int cl  = lane & 15;
  const int co0 = (np * 2) * 16 + cl;
  const int co1 = co0 + 16;
  const float inv0 = bnG[co0] / sqrtf(bnV[co0] + EPS_BN);
  const float sh0  = bnB[co0] - bnM[co0] * inv0;
  const float inv1 = bnG[co1] / sqrtf(bnV[co1] + EPS_BN);
  const float sh1  = bnB[co1] - bnM[co1] * inv1;
  const int rowBase = pixBase + hi * 8;
#pragma unroll
  for (int j = 0; j < 8; ++j) {
    const int pp = rowBase + j;
    float v0 = acc0[j] * inv0 + sh0;
    float v1 = acc1[j] * inv1 + sh1;
    if (res0) {
      v0 += res0[(size_t)pp * r0Stride + r0Off + co0];
      v1 += res0[(size_t)pp * r0Stride + r0Off + co1];
    }
    if (res1) {
      v0 += res1[(size_t)pp * r1Stride + r1Off + co0];
      v1 += res1[(size_t)pp * r1Stride + r1Off + co1];
    }
    if (doLeaky) {
      v0 = v0 >= 0.f ? v0 : 0.1f * v0;
      v1 = v1 >= 0.f ? v1 : 0.1f * v1;
    }
    if (outF) {
      outF[(size_t)pp * oStride + oOff + co0] = v0;
      outF[(size_t)pp * oStride + oOff + co1] = v1;
    }
    if (outQ) {
      const size_t qb = (size_t)qslot(pp) * qStride;
      outQ[qb + qperm(qOff + co0)] = q8(v0);
      outQ[qb + qperm(qOff + co1)] = q8(v1);
    }
  }
}

__global__ void sfe1_conv(const float* __restrict__ x, const float* __restrict__ w,
                          const float* __restrict__ bias,
                          float* __restrict__ outF, unsigned char* __restrict__ outQ) {
  int t = blockIdx.x * blockDim.x + threadIdx.x;
  if (t >= NPIX * 64) return;
  int co = t & 63; int p = t >> 6;
  int xx = p & 63, yy = (p >> 6) & 63, n = p >> 12;
  float acc = bias[co];
  for (int ci = 0; ci < 3; ++ci)
    for (int ky = 0; ky < 3; ++ky) {
      int ys = yy + ky - 1; if ((unsigned)ys >= 64u) continue;
      for (int kx = 0; kx < 3; ++kx) {
        int xs = xx + kx - 1; if ((unsigned)xs >= 64u) continue;
        acc += x[(((size_t)n * 3 + ci) * 64 + ys) * 64 + xs]
             * w[(((size_t)co * 3 + ci) * 3 + ky) * 3 + kx];
      }
    }
  outF[(size_t)p * 64 + co] = acc;
  outQ[(size_t)qslot(p) * 64 + qperm(co)] = q8(acc);
}

__global__ void copy_quant(const float* __restrict__ src, int sStride, int sOff,
                           float* __restrict__ dstF, int dStride, int dOff,
                           unsigned char* __restrict__ dstQ) {
  int t = blockIdx.x * blockDim.x + threadIdx.x;
  if (t >= NPIX * 64) return;
  int c = t & 63; int p = t >> 6;
  float v = src[(size_t)p * sStride + sOff + c];
  dstF[(size_t)p * dStride + dOff + c] = v;
  dstQ[(size_t)qslot(p) * dStride + qperm(dOff + c)] = q8(v);
}

__global__ void add3_kernel(const float* __restrict__ a, const float* __restrict__ b,
                            const float* __restrict__ c, float* __restrict__ o, int n) {
  int t = blockIdx.x * blockDim.x + threadIdx.x;
  if (t < n) o[t] = a[t] + b[t] + c[t];
}

__global__ void up_conv_shuffle(const float* __restrict__ hin, const float* __restrict__ w,
                                const float* __restrict__ bias, float* __restrict__ out) {
  int t = blockIdx.x * blockDim.x + threadIdx.x;
  if (t >= NPIX * 12) return;
  int co = t % 12; int p = t / 12;
  int xx = p & 63, yy = (p >> 6) & 63, n = p >> 12;
  float acc = bias[co];
  for (int ky = 0; ky < 3; ++ky) {
    int ys = yy + ky - 1; if ((unsigned)ys >= 64u) continue;
    for (int kx = 0; kx < 3; ++kx) {
      int xs = xx + kx - 1; if ((unsigned)xs >= 64u) continue;
      const float* hp = hin + ((size_t)((n * 64 + ys) * 64 + xs)) * 64;
      const float* wp = w + (((size_t)co * 64) * 3 + ky) * 3 + kx;
      for (int ci = 0; ci < 64; ++ci) acc += hp[ci] * wp[(size_t)ci * 9];
    }
  }
  int c = co >> 2, ry = (co >> 1) & 1, rx = co & 1;
  out[(((size_t)n * 3 + c) * 128 + (yy * 2 + ry)) * 128 + (xx * 2 + rx)] = acc;
}

extern "C" void kernel_launch(void* const* d_in, const int* in_sizes, int n_in,
                              void* d_out, int out_size, void* d_ws, size_t ws_size,
                              hipStream_t stream) {
  (void)in_sizes; (void)n_in; (void)out_size; (void)ws_size;

  int idx = 0;
  auto F = [&]() { return (const float*)d_in[idx++]; };
  const float* X       = F();
  const float* sfe1_w  = F();
  const float* sfe1_b  = F();
  const float* sfe2_w  = F();
  const float* sfe2_bn[4]; for (int i = 0; i < 4; ++i) sfe2_bn[i] = F();
  struct Rdb {
    const float* cw[8]; const float* cbn[8][4];
    const float* lffw;  const float* lffbn[4];
  } rdb[16];
  for (int r = 0; r < 16; ++r) {
    for (int c = 0; c < 8; ++c) {
      rdb[r].cw[c] = F();
      for (int i = 0; i < 4; ++i) rdb[r].cbn[c][i] = F();
    }
    rdb[r].lffw = F();
    for (int i = 0; i < 4; ++i) rdb[r].lffbn[i] = F();
  }
  const float* gff1_w  = F();
  const float* gff1_bn[4]; for (int i = 0; i < 4; ++i) gff1_bn[i] = F();
  const float* gff2_w  = F();
  const float* gff2_bn[4]; for (int i = 0; i < 4; ++i) gff2_bn[i] = F();
  const float* up_w    = F();
  const float* up_b    = F();

  uintptr_t base = (uintptr_t)d_ws;
  auto alloc = [&](size_t bytes) -> void* {
    uintptr_t a = (base + 255) & ~(uintptr_t)255;
    base = a + bytes;
    return (void*)a;
  };
  float*         F1f  = (float*)alloc((size_t)NPIX * 64 * 4);
  unsigned char* F1q  = (unsigned char*)alloc((size_t)NSLOT * 64);
  float*         Hf   = (float*)alloc((size_t)NPIX * 64 * 4);
  float*         CATf = (float*)alloc((size_t)NPIX * 576 * 4);
  unsigned char* CATq = (unsigned char*)alloc((size_t)NSLOT * 576);
  unsigned char* ROUTq= (unsigned char*)alloc((size_t)NSLOT * 1024);
  float*         Gf   = (float*)alloc((size_t)NPIX * 64 * 4);
  unsigned char* Gq   = (unsigned char*)alloc((size_t)NSLOT * 64);
  float*         G2f  = (float*)alloc((size_t)NPIX * 64 * 4);
  float*         UPin = (float*)alloc((size_t)NPIX * 64 * 4);

  auto packBytes = [](int cin, int taps) -> size_t {
    int pairs = ((cin / 64) + 1) >> 1;
    return (size_t)taps * pairs * 4 * 2048;
  };
  unsigned char* pk_sfe2 = (unsigned char*)alloc(packBytes(64, 9));
  unsigned char* pk_rdb[16][8];
  unsigned char* pk_lff[16];
  for (int r = 0; r < 16; ++r) {
    for (int c = 0; c < 8; ++c)
      pk_rdb[r][c] = (unsigned char*)alloc(packBytes(64 * (1 + c), 9));
    pk_lff[r] = (unsigned char*)alloc(packBytes(576, 1));
  }
  unsigned char* pk_gff1 = (unsigned char*)alloc(packBytes(1024, 1));
  unsigned char* pk_gff2 = (unsigned char*)alloc(packBytes(64, 9));

  auto zfill = [&](void* ptr, size_t bytes) {
    int nW = (int)(bytes / 4);
    zero_words<<<(nW + 255) / 256, 256, 0, stream>>>((unsigned int*)ptr, nW);
  };
  zfill(F1q, (size_t)NSLOT * 64);
  zfill(CATq, (size_t)NSLOT * 576);
  zfill(Gq, (size_t)NSLOT * 64);

  auto pack = [&](const float* w, unsigned char* dst, int cin, int taps) {
    int cCh = cin / 64;
    int pairs = (cCh + 1) >> 1;
    int total = taps * pairs * 4 * 2048;
    pack_weights_kernel<<<(total + 255) / 256, 256, 0, stream>>>(
        w, dst, cin, taps, cCh, pairs);
  };
  pack(sfe2_w, pk_sfe2, 64, 9);
  for (int r = 0; r < 16; ++r) {
    for (int c = 0; c < 8; ++c) pack(rdb[r].cw[c], pk_rdb[r][c], 64 * (1 + c), 9);
    pack(rdb[r].lffw, pk_lff[r], 576, 1);
  }
  pack(gff1_w, pk_gff1, 1024, 1);
  pack(gff2_w, pk_gff2, 64, 9);

  auto bgemm = [&](const unsigned char* Aq, int aStride, int cin, int taps,
                   const unsigned char* pk, const float* const bn[4],
                   const float* r0, int r0s, int r0o,
                   const float* r1, int r1s, int r1o, int leaky,
                   float* oF, int oS, int oO,
                   unsigned char* oQ, int qS, int qO) {
    if (taps == 9)
      bgemm_fp8<9><<<NPIX / 64, 256, 0, stream>>>(
          Aq, aStride, cin / 64, pk, bn[0], bn[1], bn[2], bn[3],
          r0, r0s, r0o, r1, r1s, r1o, leaky, oF, oS, oO, oQ, qS, qO);
    else
      bgemm_fp8<1><<<NPIX / 64, 256, 0, stream>>>(
          Aq, aStride, cin / 64, pk, bn[0], bn[1], bn[2], bn[3],
          r0, r0s, r0o, r1, r1s, r1o, leaky, oF, oS, oO, oQ, qS, qO);
  };

  const int EW = (NPIX * 64 + 255) / 256;

  sfe1_conv<<<EW, 256, 0, stream>>>(X, sfe1_w, sfe1_b, F1f, F1q);

  bgemm(F1q, 64, 64, 9, pk_sfe2, sfe2_bn,
        F1f, 64, 0, nullptr, 0, 0, 0,
        Hf, 64, 0, nullptr, 0, 0);

  for (int r = 0; r < 16; ++r) {
    copy_quant<<<EW, 256, 0, stream>>>(Hf, 64, 0, CATf, 576, 0, CATq);
    for (int c = 0; c < 8; ++c) {
      const int cin = 64 * (1 + c);
      const int off = 64 * (1 + c);
      bgemm(CATq, 576, cin, 9, pk_rdb[r][c], rdb[r].cbn[c],
            (c == 0 ? CATf : nullptr), 576, 0, nullptr, 0, 0, 1,
            CATf, 576, off, CATq, 576, off);
    }
    bgemm(CATq, 576, 576, 1, pk_lff[r], rdb[r].lffbn,
          CATf, 576, 512, Hf, 64, 0, 0,
          Hf, 64, 0, ROUTq, 1024, 64 * r);
  }

  bgemm(ROUTq, 1024, 1024, 1, pk_gff1, gff1_bn,
        nullptr, 0, 0, nullptr, 0, 0, 0,
        Gf, 64, 0, Gq, 64, 0);

  bgemm(Gq, 64, 64, 9, pk_gff2, gff2_bn,
        Gf, 64, 0, nullptr, 0, 0, 0,
        G2f, 64, 0, nullptr, 0, 0);

  add3_kernel<<<EW, 256, 0, stream>>>(G2f, Hf, F1f, UPin, NPIX * 64);

  up_conv_shuffle<<<(NPIX * 12 + 255) / 256, 256, 0, stream>>>(
      UPin, up_w, up_b, (float*)d_out);
}
